// QuantileDilatedDenseSAGEConv_45818711113826
// MI455X (gfx1250) — matrix operand order repaired and checked by reading, not compiled, not run on hardware
//
#include <hip/hip_runtime.h>
#include <hip/hip_bf16.h>
#include <math.h>

// ---------------------------------------------------------------------------
// QuantileDilatedDenseSAGEConv for MI455X (gfx1250, wave32)
// B=4, N=512, Fin=Fout=64.  Problem is bandwidth/latency-trivial (~6MB traffic
// vs 23.3 TB/s HBM, 192MB L2), so we keep full f32 precision and use the CDNA5
// WMMA path (V_WMMA_F32_16X16X4_F32) for the fused GEMM + bias + L2-normalize.
// ---------------------------------------------------------------------------

#define BB   4
#define NN   512
#define FF   64
#define ROWS (BB * NN)      // 2048
#define KTOT 128            // concat [agg | x]

typedef __attribute__((ext_vector_type(2)))  float    v2f;
typedef __attribute__((ext_vector_type(8)))  float    v8f;
typedef __attribute__((ext_vector_type(16))) _Float16 v16h;

#if defined(__has_builtin)
#  if __has_builtin(__builtin_amdgcn_wmma_f32_16x16x4_f32)
#    define USE_WMMA_F32 1
#  endif
#endif
#ifndef USE_WMMA_F32
#  define USE_WMMA_F32 0
#endif

// ---------------------------------------------------------------------------
// Kernel 1: per-row neighbor mask + dilation removal bits.
// One wave32 per row; lane L owns j in [16L, 16L+16).
// m    = (adj_>0) & ~eye          (adj_ has forced diag 1.0)
// num  = sum(m);  skip = num>10 ? ceil(num/2) : 1
// remove bit set where (rank % skip == skip-1)  (rank = running index over m)
// ---------------------------------------------------------------------------
__global__ __launch_bounds__(256) void k_mask(const float* __restrict__ adj,
                                              unsigned short* __restrict__ removeW,
                                              unsigned short* __restrict__ amaskW)
{
    const int lane = threadIdx.x & 31;
    const int row  = blockIdx.x * 8 + (threadIdx.x >> 5);   // 0..2047
    const int i    = row & (NN - 1);

    const float* arow = adj + (size_t)row * NN;
    const int j0 = lane * 16;

    unsigned m = 0, am = 0;
    const float4* av = (const float4*)(arow + j0);
#pragma unroll
    for (int q = 0; q < 4; ++q) {
        float4 v = av[q];
        float  e[4] = {v.x, v.y, v.z, v.w};
#pragma unroll
        for (int t = 0; t < 4; ++t) {
            const int j   = j0 + q * 4 + t;
            const int bit = q * 4 + t;
            const bool pos  = (e[t] > 0.0f);
            const bool diag = (j == i);
            if (pos || diag)  am |= (1u << bit);   // adj_ > 0 (diag forced 1)
            if (pos && !diag) m  |= (1u << bit);   // m = (adj_>0) & ~eye
        }
    }

    // wave32 inclusive scan of per-lane popcounts
    int cnt  = __popc(m);
    int scan = cnt;
#pragma unroll
    for (int off = 1; off < 32; off <<= 1) {
        int t = __shfl_up(scan, off, 32);
        if (lane >= off) scan += t;
    }
    const int num  = __shfl(scan, 31, 32);
    const int excl = scan - cnt;

    const int skip = (num > 10) ? ((num + 1) >> 1) : 1;   // T_THRESH=10, K_DIL=2

    unsigned rm = 0;
    if (skip > 1) {
        int rank = excl;
        unsigned mm = m;
        while (mm) {
            const int t = __ffs(mm) - 1;
            mm &= mm - 1;
            if (rank % skip == skip - 1) rm |= (1u << t);
            ++rank;
        }
    }
    removeW[row * 32 + lane] = (unsigned short)rm;
    amaskW [row * 32 + lane] = (unsigned short)am;
}

// ---------------------------------------------------------------------------
// Kernel 2: symmetrize removal (remove | remove^T), final neighbor mask + deg.
// ---------------------------------------------------------------------------
__global__ __launch_bounds__(256) void k_sym(const unsigned short* __restrict__ removeW,
                                             const unsigned short* __restrict__ amaskW,
                                             unsigned short* __restrict__ nbrW,
                                             int* __restrict__ deg)
{
    const int lane = threadIdx.x & 31;
    const int row  = blockIdx.x * 8 + (threadIdx.x >> 5);
    const int b    = row >> 9;
    const int i    = row & (NN - 1);

    unsigned nb = (unsigned)amaskW[row * 32 + lane] & ~(unsigned)removeW[row * 32 + lane] & 0xFFFFu;

    const int wsel = i >> 4;
    const int bsel = i & 15;
    unsigned mrem = nb;
    while (mrem) {
        const int t = __ffs(mrem) - 1;
        mrem &= mrem - 1;
        const int j = lane * 16 + t;
        const unsigned rc = removeW[((size_t)(b * NN + j)) * 32 + wsel];
        if ((rc >> bsel) & 1u) nb &= ~(1u << t);           // remove^T bit
    }
    nbrW[row * 32 + lane] = (unsigned short)nb;

    int c = __popc(nb);
#pragma unroll
    for (int off = 16; off > 0; off >>= 1) c += __shfl_down(c, off, 32);
    if (lane == 0) deg[row] = c;
}

// ---------------------------------------------------------------------------
// Kernel 3: per-row, per-channel quantile aggregation.
// Block = 256 threads per row. Dynamic LDS: vals[512*64] f32 (128KB, fits in
// CDNA5's 320KB/WGP) + index list + counter. Cooperative bitonic sort along
// the (pow2-padded, inf-filled) neighbor axis, independently per channel.
// ---------------------------------------------------------------------------
__global__ __launch_bounds__(256) void k_quantile(const float* __restrict__ x,
                                                  const unsigned short* __restrict__ nbrW,
                                                  float* __restrict__ agg)
{
    extern __shared__ char smem_raw[];
    float*          vals = (float*)smem_raw;                              // 512*64 f32
    unsigned short* nidx = (unsigned short*)(smem_raw + NN * FF * 4);     // 512 u16
    int*            pcnt = (int*)(smem_raw + NN * FF * 4 + NN * 2);       // 1 int

    const int tid = threadIdx.x;
    const int row = blockIdx.x;
    const int b   = row >> 9;

    if (tid == 0) pcnt[0] = 0;
    __syncthreads();

    if (tid < 32) {
        unsigned w = nbrW[row * 32 + tid];
        while (w) {
            const int t = __ffs(w) - 1;
            w &= w - 1;
            const int slot = atomicAdd(&pcnt[0], 1);
            nidx[slot] = (unsigned short)(tid * 16 + t);
        }
    }
    __syncthreads();
    const int degv = pcnt[0];

    int P = 1;
    while (P < degv) P <<= 1;                      // pow2 pad (P>=1)
    const int total = P * FF;

    // gather: consecutive tid -> consecutive channel f (coalesced 256B/neighbor)
    for (int e = tid; e < total; e += 256) {
        const int n = e >> 6, f = e & 63;
        float v = __builtin_inff();
        if (n < degv)
            v = x[((size_t)(b * NN) + nidx[n]) * FF + f];
        vals[e] = v;
    }

    // bitonic sort ascending along n, per channel
    for (int k = 2; k <= P; k <<= 1) {
        for (int j = k >> 1; j > 0; j >>= 1) {
            __syncthreads();
            for (int e = tid; e < total; e += 256) {
                const int n = e >> 6, f = e & 63;
                const int l = n ^ j;
                if (l > n) {
                    const bool up = ((n & k) == 0);
                    const float a = vals[n * 64 + f];
                    const float c = vals[l * 64 + f];
                    if (up ? (a > c) : (a < c)) {
                        vals[n * 64 + f] = c;
                        vals[l * 64 + f] = a;
                    }
                }
            }
        }
    }
    __syncthreads();

    if (tid < FF) {
        float aggv = 0.0f;
        if (degv > 0) {
            const float degm1 = fmaxf((float)degv - 1.0f, 0.0f);
            const float taus[3] = {0.25f, 0.5f, 0.75f};
            const float wq[3]   = {0.25f, 0.5f, 0.25f};
#pragma unroll
            for (int t = 0; t < 3; ++t) {
                const float pos = taus[t] * degm1;
                const int lo = (int)floorf(pos);
                const int hi = (int)ceilf(pos);
                const float fr = pos - (float)lo;
                const float qlo = vals[lo * 64 + tid];
                const float qhi = vals[hi * 64 + tid];
                aggv += wq[t] * (qlo * (1.0f - fr) + qhi * fr);
            }
        }
        agg[(size_t)row * FF + tid] = aggv;
    }
}

// ---------------------------------------------------------------------------
// Kernel 4: out = [agg|x] @ [W_l;W_r] + b_l, then row L2-normalize.
// Block = 128 threads (4 wave32s), 16 rows per block; each wave owns one
// 16x16 output tile (Fout split in 4). Full f32 via V_WMMA_F32_16X16X4_F32
// (32 K-steps of 4), falling back to the codegen-confirmed f16 WMMA if the
// f32 builtin is unavailable.
// ---------------------------------------------------------------------------
__global__ __launch_bounds__(128) void k_gemm_norm(const float* __restrict__ agg,
                                                   const float* __restrict__ x,
                                                   const float* __restrict__ Wl,
                                                   const float* __restrict__ bl,
                                                   const float* __restrict__ Wr,
                                                   float* __restrict__ out)
{
    __shared__ float As[16 * KTOT];      // 8 KB  A tile (16 rows x 128)
    __shared__ float Ws[KTOT * FF];      // 32 KB stacked weights (128 x 64)
    __shared__ float Os[16 * FF];        // 4 KB  output tile
    __shared__ float Nrm[16];
#if !USE_WMMA_F32
    __shared__ _Float16 Ah[16 * KTOT];
    __shared__ _Float16 Wh[KTOT * FF];
#endif

    const int tid = threadIdx.x;
    const int r0  = blockIdx.x * 16;

    for (int e = tid; e < KTOT * FF; e += 128) {
        const int k = e >> 6, c = e & 63;
        const float w = (k < FF) ? Wl[k * FF + c] : Wr[(k - FF) * FF + c];
        Ws[e] = w;
#if !USE_WMMA_F32
        Wh[e] = (_Float16)w;
#endif
    }
    for (int e = tid; e < 16 * KTOT; e += 128) {
        const int m = e >> 7, k = e & (KTOT - 1);
        const int row = r0 + m;
        const float a = (k < FF) ? agg[(size_t)row * FF + k]
                                 : x[(size_t)row * FF + (k - FF)];
        As[e] = a;
#if !USE_WMMA_F32
        Ah[e] = (_Float16)a;
#endif
    }
    __syncthreads();

    const int wave    = tid >> 5;
    const int lane    = tid & 31;
    const int colBase = wave * 16;
    const int M       = lane & 15;
    const int hiHalf  = lane >> 4;

    v8f acc = {};
#if USE_WMMA_F32
    // A frag (ISA 7.12.2, 32-bit 16x4): VGPR v -> K = v + 2*hiHalf, row M=lane&15
    // B frag mirrored: VGPR v -> K = v + 2*hiHalf, col N=lane&15
    const int koff = 2 * hiHalf;
#pragma unroll
    for (int kk = 0; kk < KTOT / 4; ++kk) {
        const int kb = kk * 4 + koff;
        v2f a, b;
        a[0] = As[M * KTOT + kb];
        a[1] = As[M * KTOT + kb + 1];
        b[0] = Ws[(kb)     * FF + colBase + M];
        b[1] = Ws[(kb + 1) * FF + colBase + M];
        acc = __builtin_amdgcn_wmma_f32_16x16x4_f32(false, a, false, b,
                                                    (short)0, acc, false, false);
    }
#else
    // f16 fallback (codegen-confirmed builtin); K=128 -> 4 steps of 32
#pragma unroll
    for (int kk = 0; kk < KTOT / 32; ++kk) {
        v16h a, b;
#pragma unroll
        for (int h = 0; h < 16; ++h) {
            const int ka = kk * 32 + ((h < 8 ? h : h + 8) + hiHalf * 8);
            const int kb = kk * 32 + ((h < 8 ? h : h + 8) + hiHalf * 8);
            a[h] = Ah[M * KTOT + ka];
            b[h] = Wh[kb * FF + colBase + M];
        }
        acc = __builtin_amdgcn_wmma_f32_16x16x32_f16(false, a, false, b,
                                                     (short)0, acc, false, false);
    }
#endif

    // C/D layout: VGPR r -> (M = r + 8*hiHalf, N = lane&15)
#pragma unroll
    for (int r = 0; r < 8; ++r) {
        const int m = r + 8 * hiHalf;
        const int c = colBase + M;
        Os[m * FF + c] = acc[r] + bl[c];
    }
    __syncthreads();

    if (tid < 16) {
        float s = 0.0f;
#pragma unroll 8
        for (int c = 0; c < FF; ++c) {
            const float v = Os[tid * FF + c];
            s += v * v;
        }
        Nrm[tid] = fmaxf(sqrtf(s), 1e-12f);
    }
    __syncthreads();

    for (int e = tid; e < 16 * FF; e += 128) {
        const int m = e >> 6;
        out[(size_t)(r0 + m) * FF + (e & 63)] = Os[e] / Nrm[m];
    }
}

// ---------------------------------------------------------------------------
// Launch. Inputs: x(4,512,64) f32, adj(4,512,512) f32, W_l(64,64), b_l(64),
// W_r(64,64). Output: (4,512,64) f32. Workspace: ~926 KB.
// ---------------------------------------------------------------------------
extern "C" void kernel_launch(void* const* d_in, const int* in_sizes, int n_in,
                              void* d_out, int out_size, void* d_ws, size_t ws_size,
                              hipStream_t stream)
{
    (void)in_sizes; (void)n_in; (void)out_size; (void)ws_size;

    const float* x   = (const float*)d_in[0];
    const float* adj = (const float*)d_in[1];
    const float* Wl  = (const float*)d_in[2];
    const float* bl  = (const float*)d_in[3];
    const float* Wr  = (const float*)d_in[4];
    float*       out = (float*)d_out;

    char* ws = (char*)d_ws;
    unsigned short* removeW = (unsigned short*)(ws);                 // 128 KB
    unsigned short* amaskW  = (unsigned short*)(ws + 131072);        // 128 KB
    unsigned short* nbrW    = (unsigned short*)(ws + 262144);        // 128 KB
    int*            deg     = (int*)(ws + 393216);                   // 8 KB
    float*          agg     = (float*)(ws + 401408);                 // 512 KB

    // 1 wave per row, 8 waves per block -> 256 blocks of 256 threads
    k_mask<<<ROWS / 8, 256, 0, stream>>>(adj, removeW, amaskW);
    k_sym <<<ROWS / 8, 256, 0, stream>>>(removeW, amaskW, nbrW, deg);

    // 1 block per row; dynamic LDS sized for worst-case deg=512
    const size_t smem = (size_t)NN * FF * 4 + (size_t)NN * 2 + 16;   // 132112 B
    k_quantile<<<ROWS, 256, smem, stream>>>(x, nbrW, agg);

    (void)deg; // deg folded into k_quantile's LDS count (identical by construction)

    // 16 rows per block -> 128 blocks of 128 threads (4 waves)
    k_gemm_norm<<<ROWS / 16, 128, 0, stream>>>(agg, x, Wl, bl, Wr, out);
}
